// VAE_60936995995662
// MI455X (gfx1250) — compile-verified, matrix operand order checked by reading, not run on hardware
//
#include <hip/hip_runtime.h>
#include <math.h>

typedef __attribute__((ext_vector_type(16))) _Float16     v16h;
typedef __attribute__((ext_vector_type(8)))  _Float16     v8h;
typedef __attribute__((ext_vector_type(8)))  float        v8f;
typedef __attribute__((ext_vector_type(4)))  unsigned int v4u;
typedef __attribute__((ext_vector_type(4)))  int          v4i;
typedef __attribute__((ext_vector_type(8)))  int          v8i;

#define BM 128
#define BN 128
#define BK 32
#define LDA 40                    // halves; 80B row stride (64B data + 16B TDM pad)
#define LDSBUF (BM * LDA * 2)     // bytes per tile buffer

// Tensor Data Mover: load one 128x32-half 2D tile into LDS with row padding.
// clang-22 (ROCm 7.2): 5-arg builtin; clang-23 (amdgpu-toolchain): 6-arg.
#if defined(__clang_major__) && (__clang_major__ >= 23)
#define TDM_LOAD(g0, g1) __builtin_amdgcn_tensor_load_to_lds((g0), (g1), \
    (v4i){0,0,0,0}, (v4i){0,0,0,0}, (v8i){0,0,0,0,0,0,0,0}, 0)
#else
#define TDM_LOAD(g0, g1) __builtin_amdgcn_tensor_load_to_lds((g0), (g1), \
    (v4i){0,0,0,0}, (v4i){0,0,0,0}, 0)
#endif

__device__ __forceinline__ int koff_of(int h, int lhi) {
    // ISA 16-bit A/B 16x32 layout: lanes 0-15 halves 0-7 -> K0-7, 8-15 -> K16-23;
    // lanes 16-31: +8
    return (h & 7) + ((h >> 3) << 4) + (lhi << 3);
}

// O = act(Ah[MxK] @ Wh[NpadxK]^T + b). f16 operands, K%32==0, W zero-padded rows.
// Tiles staged by TDM (tensor_load_to_lds), double-buffered, consumed by WMMA.
// mode 0: tanh -> f16 Oh[MxN]
// mode 1: (mean | clamp logvar | approx) -> f32, guard col<192
// mode 3: sigmoid -> f32 O0[MxN], guard col<N
__global__ __launch_bounds__(256)
void gemm_f16_wmma(const _Float16* __restrict__ Ah,
                   const _Float16* __restrict__ Wh,
                   const float* __restrict__ b0, const float* __restrict__ b1,
                   const float* __restrict__ b2,
                   _Float16* __restrict__ Oh,
                   float* __restrict__ O0, float* __restrict__ O1,
                   float* __restrict__ O2,
                   int N, int K, int mode)
{
    __shared__ __align__(16) _Float16 As[2][BM * LDA];
    __shared__ __align__(16) _Float16 Bs[2][BM * LDA];

    const int t    = threadIdx.x;
    const int wave = t >> 5;
    const int lane = t & 31;
    const int lrow = lane & 15;
    const int lhi  = lane >> 4;
    const int rowBase = blockIdx.x * BM;
    const int colBase = blockIdx.y * BN;
    const int wm = wave >> 1;   // 0..3 -> 32-row band
    const int wn = wave & 1;    // 0..1 -> 64-col band

    // ---- TDM descriptor group1 (same for A and B tiles): 2D tensor, f16,
    // tensor_dim0=K, tensor_dim1=128 (block tile rows), tile 32x128,
    // dim0_stride=K, LDS pad: after 16 dwords (64B row) insert 4 dwords (16B).
    const unsigned d0 = (1u << 16)              // data_size = 1 -> 2 bytes
                      | (1u << 20)              // pad_enable
                      | (3u << 22)              // pad_interval: 16 dwords
                      | (3u << 25);             // pad_amount:   4 dwords
    const unsigned d1 = ((unsigned)K & 0xFFFFu) << 16;                 // tensor_dim0 lo
    const unsigned d2 = ((unsigned)K >> 16) | (128u << 16);            // dim0 hi | tensor_dim1 lo
    const unsigned d3 = (32u << 16);                                   // dim1 hi(0) | tile_dim0=32
    const unsigned d4 = 128u;                                          // tile_dim1=128, tile_dim2=0
    const unsigned d5 = (unsigned)K;                                   // dim0_stride lo
    const v8i g1 = { (int)d0, (int)d1, (int)d2, (int)d3,
                     (int)d4, (int)d5, 0, 0 };

    const _Float16* Abase = Ah + (size_t)rowBase * K;
    const _Float16* Bbase = Wh + (size_t)colBase * K;
    const unsigned ldsA = (unsigned)(uintptr_t)(&As[0][0]);  // LDS byte offset
    const unsigned ldsB = (unsigned)(uintptr_t)(&Bs[0][0]);

    auto tdm = [&](const _Float16* gptr, unsigned lds_byte) {
        const unsigned long long ga = (unsigned long long)(uintptr_t)gptr;
        v4u g0 = { 1u,                                   // count=1, user descriptor
                   lds_byte,                             // lds_addr
                   (unsigned)ga,                         // global_addr[31:0]
                   ((unsigned)(ga >> 32) & 0x01FFFFFFu)  // global_addr[56:32]
                     | 0x80000000u };                    // type=2 (image/tensor)
        TDM_LOAD(g0, g1);
    };

    v8f acc[2][4];
#pragma unroll
    for (int mi = 0; mi < 2; ++mi)
#pragma unroll
        for (int ni = 0; ni < 4; ++ni)
            acc[mi][ni] = (v8f){0.f,0.f,0.f,0.f,0.f,0.f,0.f,0.f};

    const int steps = K / BK;
    // prologue: k-step 0 into buffer 0 (wave0 -> A tile, wave1 -> B tile)
    if (wave == 0)      tdm(Abase, ldsA);
    else if (wave == 1) tdm(Bbase, ldsB);

    for (int s = 0; s < steps; ++s) {
        const int cur = s & 1;
        // prefetch next k-step into other buffer, then drain current tile
        if (wave == 0) {
            if (s + 1 < steps) {
                tdm(Abase + (size_t)(s + 1) * BK, ldsA + (1 - cur) * LDSBUF);
                __builtin_amdgcn_s_wait_tensorcnt(1);
            } else {
                __builtin_amdgcn_s_wait_tensorcnt(0);
            }
        } else if (wave == 1) {
            if (s + 1 < steps) {
                tdm(Bbase + (size_t)(s + 1) * BK, ldsB + (1 - cur) * LDSBUF);
                __builtin_amdgcn_s_wait_tensorcnt(1);
            } else {
                __builtin_amdgcn_s_wait_tensorcnt(0);
            }
        }
        __syncthreads();   // tile[cur] complete + prior readers of tile[1-cur] done

        v16h af[2], bf[4];
#pragma unroll
        for (int mi = 0; mi < 2; ++mi) {
            const int r = (wm * 32 + mi * 16 + lrow) * LDA;
#pragma unroll
            for (int h = 0; h < 16; ++h)
                af[mi][h] = As[cur][r + koff_of(h, lhi)];
        }
#pragma unroll
        for (int ni = 0; ni < 4; ++ni) {
            const int r = (wn * 64 + ni * 16 + lrow) * LDA;
#pragma unroll
            for (int h = 0; h < 16; ++h)
                bf[ni][h] = Bs[cur][r + koff_of(h, lhi)];
        }
#pragma unroll
        for (int mi = 0; mi < 2; ++mi)
#pragma unroll
            for (int ni = 0; ni < 4; ++ni)
                acc[mi][ni] = __builtin_amdgcn_wmma_f32_16x16x32_f16(
                    false, af[mi], false, bf[ni],
                    (short)0, acc[mi][ni], false, false);
    }

    // Epilogue. C/D layout: VGPR r -> row r + 8*lhi, col = lane&15.
#pragma unroll
    for (int mi = 0; mi < 2; ++mi) {
#pragma unroll
        for (int ni = 0; ni < 4; ++ni) {
            const int col = colBase + wn * 64 + ni * 16 + lrow;
#pragma unroll
            for (int r = 0; r < 8; ++r) {
                const int row = rowBase + wm * 32 + mi * 16 + r + lhi * 8;
                float v = acc[mi][ni][r];
                if (mode == 0) {
                    v = tanhf(v + b0[col]);
                    Oh[(size_t)row * N + col] = (_Float16)v;
                } else if (mode == 1) {
                    if (col < 192) {
                        const int sub = col >> 6, c = col & 63;
                        if (sub == 0)      O0[(size_t)row * 64 + c] = v + b0[c];
                        else if (sub == 1) O1[(size_t)row * 64 + c] =
                                 fminf(fmaxf(v + b1[c], -4.5f), 0.0f);
                        else               O2[(size_t)row * 64 + c] = v + b2[c];
                    }
                } else { // mode 3
                    if (col < N)
                        O0[(size_t)row * N + col] = 1.0f / (1.0f + __expf(-(v + b0[col])));
                }
            }
        }
    }
}

// f32 -> f16 with zero padding in both dims. grid: (ceil(Kp/256), rowsPad)
__global__ void cvt_f16_pad(const float* __restrict__ src, _Float16* __restrict__ dst,
                            int rows, int K, int Kp)
{
    const int r = blockIdx.y;
    const int k = blockIdx.x * 256 + threadIdx.x;
    if (k >= Kp) return;
    float v = (r < rows && k < K) ? src[(size_t)r * K + k] : 0.0f;
    dst[(size_t)r * Kp + k] = (_Float16)v;
}

// Rank-1 Gaussian reparameterization (Sherman-Morrison + analytic Cholesky of
// diag(e) - s vv^T): r_j = 1 + suffix(e a^2), Ljj = sqrt(e r_{j+1}/r_j),
// b_j = -e a/(r_j Ljj), z_j = mean_j + Ljj eps_j + b_j suffix_{k>j}(e a eps).
__global__ void vae_reparam(const float* __restrict__ mean,
                            const float* __restrict__ logvar,
                            const float* __restrict__ approx,
                            const float* __restrict__ eps,
                            float* __restrict__ z, _Float16* __restrict__ zh, int B)
{
    const int s = blockIdx.x * blockDim.x + threadIdx.x;
    if (s >= B) return;
    const int L = 64;
    const float* mrow  = mean   + (size_t)s * L;
    const float* lvrow = logvar + (size_t)s * L;   // already clamped by GEMM2 epilogue
    const float* arow  = approx + (size_t)s * L;
    const float* erow  = eps    + (size_t)s * L;
    float*       zrow  = z      + (size_t)s * L;
    _Float16*    zhrow = zh     + (size_t)s * L;

    float r = 1.0f, S = 0.0f;
    for (int j = L - 1; j >= 0; --j) {
        const float e  = __expf(-lvrow[j]);
        const float a  = arow[j];
        const float g  = e * a * a;
        const float rn = r;          // r_{j+1}
        r += g;                      // r_j
        const float Ljj = sqrtf(e * rn / r);
        const float b   = -e * a / (r * Ljj);
        const float ep  = erow[j];
        const float zj  = mrow[j] + Ljj * ep + b * S;
        zrow[j]  = zj;
        zhrow[j] = (_Float16)zj;
        S += e * a * ep;
    }
}

extern "C" void kernel_launch(void* const* d_in, const int* in_sizes, int n_in,
                              void* d_out, int out_size, void* d_ws, size_t ws_size,
                              hipStream_t stream) {
    (void)in_sizes; (void)n_in; (void)out_size; (void)ws_size;
    const float* x    = (const float*)d_in[0];
    const float* eps  = (const float*)d_in[1];
    const float* W_ih = (const float*)d_in[2];
    const float* b_ih = (const float*)d_in[3];
    const float* W_m  = (const float*)d_in[4];
    const float* b_m  = (const float*)d_in[5];
    const float* W_lv = (const float*)d_in[6];
    const float* b_lv = (const float*)d_in[7];
    const float* W_ap = (const float*)d_in[8];
    const float* b_ap = (const float*)d_in[9];
    const float* W_lh = (const float*)d_in[10];
    const float* b_lh = (const float*)d_in[11];
    const float* W_ho = (const float*)d_in[12];
    const float* b_ho = (const float*)d_in[13];

    const int B = 16384, D = 784, H = 1024, L = 64;
    const int Dp = 800;   // 784 padded to %32
    const int NhoPad = 896, NcatPad = 256;

    // d_out: x_mean[B,784] | z[B,64] | mean | logvar | approx
    float* out    = (float*)d_out;
    float* x_mean = out;
    float* z      = x_mean + (size_t)B * D;
    float* mean   = z      + (size_t)B * L;
    float* logvar = mean   + (size_t)B * L;
    float* approx = logvar + (size_t)B * L;

    // ws carve (all f16, region sizes multiples of 8 halves -> 16B aligned)
    _Float16* w = (_Float16*)d_ws;
    _Float16* xh      = w;                    w += (size_t)B * Dp;        // 16384x800
    _Float16* Wih_h   = w;                    w += (size_t)H * Dp;        // 1024x800
    _Float16* Wcat_h  = w;                    w += (size_t)NcatPad * H;   // 256x1024
    _Float16* Wlh_h   = w;                    w += (size_t)H * L;         // 1024x64
    _Float16* Who_h   = w;                    w += (size_t)NhoPad * H;    // 896x1024
    _Float16* hidden_h= w;                    w += (size_t)B * H;         // 16384x1024
    _Float16* h2_h    = w;                    w += (size_t)B * H;         // 16384x1024
    _Float16* zh      = w;                                                 // 16384x64

    dim3 blk(256);
    // --- one-time f32->f16 conversions with padding ---
    cvt_f16_pad<<<dim3((Dp + 255) / 256, B), blk, 0, stream>>>(x, xh, B, D, Dp);
    cvt_f16_pad<<<dim3((Dp + 255) / 256, H), blk, 0, stream>>>(W_ih, Wih_h, H, D, Dp);
    cvt_f16_pad<<<dim3((H + 255) / 256, 64), blk, 0, stream>>>(W_m,  Wcat_h,            64, H, H);
    cvt_f16_pad<<<dim3((H + 255) / 256, 64), blk, 0, stream>>>(W_lv, Wcat_h + 64 * H,   64, H, H);
    cvt_f16_pad<<<dim3((H + 255) / 256, 64), blk, 0, stream>>>(W_ap, Wcat_h + 128 * H,  64, H, H);
    cvt_f16_pad<<<dim3((H + 255) / 256, 64), blk, 0, stream>>>(W_m,  Wcat_h + 192 * H,   0, H, H); // zero rows
    cvt_f16_pad<<<dim3((L + 255) / 256, H), blk, 0, stream>>>(W_lh, Wlh_h, H, L, L);
    cvt_f16_pad<<<dim3((H + 255) / 256, NhoPad), blk, 0, stream>>>(W_ho, Who_h, D, H, H);

    // 1) hidden = tanh(x @ W_ih^T + b_ih)   [K=800, N=1024] -> f16
    gemm_f16_wmma<<<dim3(B / BM, H / BN), blk, 0, stream>>>(
        xh, Wih_h, b_ih, nullptr, nullptr,
        hidden_h, nullptr, nullptr, nullptr, H, Dp, 0);
    // 2) mean | clamped logvar | approx     [K=1024, Npad=256]
    gemm_f16_wmma<<<dim3(B / BM, NcatPad / BN), blk, 0, stream>>>(
        hidden_h, Wcat_h, b_m, b_lv, b_ap,
        nullptr, mean, logvar, approx, 192, H, 1);
    // 3) z = mean + chol(cov)^T eps (closed form), also emit f16 copy
    vae_reparam<<<dim3(B / 256), blk, 0, stream>>>(mean, logvar, approx, eps, z, zh, B);
    // 4) h2 = tanh(z @ W_lh^T + b_lh)       [K=64, N=1024] -> f16
    gemm_f16_wmma<<<dim3(B / BM, H / BN), blk, 0, stream>>>(
        zh, Wlh_h, b_lh, nullptr, nullptr,
        h2_h, nullptr, nullptr, nullptr, H, L, 0);
    // 5) x_mean = sigmoid(h2 @ W_ho^T + b_ho) [K=1024, Npad=896, N=784]
    gemm_f16_wmma<<<dim3(B / BM, NhoPad / BN), blk, 0, stream>>>(
        h2_h, Who_h, b_ho, nullptr, nullptr,
        nullptr, x_mean, nullptr, nullptr, D, H, 3);
}
